// AttentionalPropagation_44633300140733
// MI455X (gfx1250) — matrix operand order repaired and checked by reading, not compiled, not run on hardware
//
#include <hip/hip_runtime.h>
#include <hip/hip_bf16.h>

// ---------------------------------------------------------------------------
// AttentionalPropagation (SuperGlue-style), D=256, NQ=NK=512, fp32 in/out.
// Factorization: aW1 @ [qb;kb;dist] = U[:,q] + V[:,k] + W1d@dist[:,p]
//   -> big GEMM shrinks from K=768 to K=256 (34 GFLOP against the mandatory
//      0.27 GB dist read = ~12us at 23.3 TB/s HBM).
// Main kernel: bf16 WMMA 16x16x32, W1d resident in VGPRs, fused
//   relu + aW2-dot epilogue producing scores directly (h never hits memory).
// Side kernels tiled 4 output rows/block to cut L2 activation re-reads 4x.
// ---------------------------------------------------------------------------

#define DD    256
#define NQ    512
#define NK    512
#define NPAIR (NQ * NK)   // 262144

typedef __bf16 bf16_t;
typedef bf16_t v16bf __attribute__((ext_vector_type(16)));
typedef float  v8f   __attribute__((ext_vector_type(8)));

// -------------------------- A fragment (W1d, bf16) -------------------------
// 16-bit A-matrix 16x32 lane layout (ISA 7.12.2):
//   lanes 0-15 (hi=0): row M=lane, K = {0..7, 16..23}
//   lanes 16-31 (hi=1): row M=lane-16, K = {8..15, 24..31}
__device__ __forceinline__ v16bf load_A_frag(const float* __restrict__ W1d,
                                             int j, int kc, int hi) {
    const float* base = W1d + j * 768 + kc * 32 + hi * 8;  // row stride 768 (aW1)
    v16bf a;
#pragma unroll
    for (int e = 0; e < 8; ++e) a[e] = (bf16_t)base[e];
#pragma unroll
    for (int e = 0; e < 8; ++e) a[8 + e] = (bf16_t)base[16 + e];
    return a;
}

// -------------------------- B fragment (dist, bf16) ------------------------
// 16-bit B-matrix 32x16 lane layout: lane = column, K = {8hi..8hi+7, 16+8hi..16+8hi+7}.
// dist is [c][p] with p-stride 1 -> each load is 64B-coalesced per half-wave;
// every dist dword is read exactly once kernel-wide.
__device__ __forceinline__ v16bf load_B_frag(const float* __restrict__ dist,
                                             int p, int kc, int hi) {
    const float* base = dist + (size_t)(kc * 32 + hi * 8) * NPAIR + p;
    v16bf b;
#pragma unroll
    for (int e = 0; e < 16; ++e) b[e] = (bf16_t)base[(size_t)(e + (e & 8)) * NPAIR];
    return b;
}

// ------------------- Kernel A: U = W1q@x + b1, V = W1k@source --------------
// 4 output rows per block: activations read once, 4 scalar weights each.
__global__ __launch_bounds__(256) void uv_kernel(
    const float* __restrict__ aW1, const float* __restrict__ ab1,
    const float* __restrict__ x, const float* __restrict__ src,
    float* __restrict__ U, float* __restrict__ V) {
    const int b    = blockIdx.x;         // 0..63 -> U rows, 64..127 -> V rows
    const bool isV = b >= 64;
    const int j0   = (isV ? b - 64 : b) * 4;
    const float* in  = isV ? src : x;
    float* out       = isV ? V : U;
    const int coff   = isV ? DD : 0;
    const int tq = threadIdx.x;          // q = tq and tq+256
    float a0[4], a1[4];
#pragma unroll
    for (int r = 0; r < 4; ++r) {
        const float bias = isV ? 0.0f : ab1[j0 + r];
        a0[r] = bias; a1[r] = bias;
    }
#pragma unroll 2
    for (int c = 0; c < DD; ++c) {
        const float xv0 = in[c * NQ + tq];
        const float xv1 = in[c * NQ + tq + 256];
#pragma unroll
        for (int r = 0; r < 4; ++r) {
            const float wv = aW1[(j0 + r) * 768 + coff + c];  // wave-uniform
            a0[r] += wv * xv0;
            a1[r] += wv * xv1;
        }
    }
#pragma unroll
    for (int r = 0; r < 4; ++r) {
        out[(j0 + r) * NQ + tq]       = a0[r];
        out[(j0 + r) * NQ + tq + 256] = a1[r];
    }
}

// ---------------- Kernel B: scores via bf16 WMMA + fused epilogue ----------
// Grid-stride over 4096 tiles of 64 pairs (one q, 64 consecutive k each).
// 8 waves/WG; wave w owns j-tiles [32w,32w+16) and [32w+16,32w+32).
// Per 16-pair group: 8 K-chunks x 2 WMMAs, then relu + aW2-dot epilogue;
// group results buffered in regs -> only 2 barriers per 64-pair tile.
__global__ __launch_bounds__(256) void mlp_scores_kernel(
    const float* __restrict__ aW1, const float* __restrict__ aW2,
    const float* __restrict__ ab2, const float* __restrict__ dist,
    const float* __restrict__ U, const float* __restrict__ V,
    float* __restrict__ scores) {
    const int tid  = threadIdx.x;
    const int lane = tid & 31;
    const int wave = tid >> 5;           // 0..7
    const int lo   = lane & 15;
    const int hi   = lane >> 4;          // 0 or 1
    const int jb0  = wave * 32;
    const int jb1  = wave * 32 + 16;

    // W1d block of aW1: columns 512..767
    const float* W1d = aW1 + 512;

    // Pre-pack all A fragments for this wave's two j-tiles: 128 VGPRs, resident.
    v16bf A0[8], A1[8];
#pragma unroll
    for (int kc = 0; kc < 8; ++kc) {
        A0[kc] = load_A_frag(W1d, jb0 + lo, kc, hi);
        A1[kc] = load_A_frag(W1d, jb1 + lo, kc, hi);
    }

    // C-tile rows owned by this lane: M = 8*hi + r  (r = accumulator index)
    const int jr0 = jb0 + 8 * hi;
    const int jr1 = jb1 + 8 * hi;
    float a2v0[8], a2v1[8];
#pragma unroll
    for (int r = 0; r < 8; ++r) {
        a2v0[r] = aW2[jr0 + r];
        a2v1[r] = aW2[jr1 + r];
    }
    const float bias2 = ab2[0];

    __shared__ float sbuf[8][4][16];

    const int tiles = NPAIR / 64;        // 4096
    for (int t = blockIdx.x; t < tiles; t += gridDim.x) {
        const int p0 = t * 64;
        const int q  = p0 >> 9;          // p0 / NK
        const int k0 = p0 & (NK - 1);

        // U column is fixed for the whole tile (one q)
        float uv0[8], uv1[8];
#pragma unroll
        for (int r = 0; r < 8; ++r) {
            uv0[r] = U[(jr0 + r) * NQ + q];
            uv1[r] = U[(jr1 + r) * NQ + q];
        }

        float tots[4];
#pragma unroll 1
        for (int g = 0; g < 4; ++g) {
            const int p = p0 + g * 16 + lo;   // this lane's output column (pair)
            const int k = k0 + g * 16 + lo;

            v8f acc0 = {};
            v8f acc1 = {};
#pragma unroll
            for (int kc = 0; kc < 8; ++kc) {
                const v16bf bfrag = load_B_frag(dist, p, kc, hi);
                acc0 = __builtin_amdgcn_wmma_f32_16x16x32_bf16(
                    false, A0[kc], false, bfrag, (short)0, acc0, false, false);
                acc1 = __builtin_amdgcn_wmma_f32_16x16x32_bf16(
                    false, A1[kc], false, bfrag, (short)0, acc1, false, false);
            }

            // Fused epilogue: h = relu(acc + U + V); partial score = a2 . h
            float part = 0.0f;
#pragma unroll
            for (int r = 0; r < 8; ++r) {
                const float h0 = acc0[r] + uv0[r] + V[(jr0 + r) * NK + k];
                const float h1 = acc1[r] + uv1[r] + V[(jr1 + r) * NK + k];
                part += a2v0[r] * fmaxf(h0, 0.0f) + a2v1[r] * fmaxf(h1, 0.0f);
            }
            // lanes n and n+16 hold disjoint row halves of the same column
            tots[g] = part + __shfl_xor(part, 16, 32);
        }

        __syncthreads();
        if (hi == 0) {
#pragma unroll
            for (int g = 0; g < 4; ++g) sbuf[wave][g][lo] = tots[g];
        }
        __syncthreads();
        if (wave < 4 && hi == 0) {          // wave w reduces group g == w
            float s = bias2;
#pragma unroll
            for (int w2 = 0; w2 < 8; ++w2) s += sbuf[w2][wave][lo];
            scores[q * NK + (k0 + wave * 16 + lo)] = s;
        }
    }
}

// ------------------- Kernel C0: global min of scores, -20 ------------------
__global__ __launch_bounds__(1024) void min_reduce_kernel(
    const float* __restrict__ scores, float* __restrict__ minout) {
    __shared__ float red[1024];
    float m = 1e30f;
    for (int i = threadIdx.x; i < NPAIR; i += 1024) m = fminf(m, scores[i]);
    red[threadIdx.x] = m;
    __syncthreads();
    for (int s = 512; s > 0; s >>= 1) {
        if (threadIdx.x < s)
            red[threadIdx.x] = fminf(red[threadIdx.x], red[threadIdx.x + s]);
        __syncthreads();
    }
    if (threadIdx.x == 0) *minout = red[0] - 20.0f;
}

// ------------- Kernel C: masked softmax (4 rows/block) + message GEMV ------
__global__ __launch_bounds__(256) void softmax_message_kernel(
    const float* __restrict__ scores, const unsigned char* __restrict__ mask,
    const float* __restrict__ negp, const float* __restrict__ source,
    float* __restrict__ message) {
    __shared__ float prob[4][NK];
    __shared__ float red[4][64];
    const int q0  = blockIdx.x * 4;
    const int tid = threadIdx.x;
    const int r   = tid >> 6;            // 0..3 : softmax row
    const int c64 = tid & 63;            // 0..63 : lane within row
    const int q   = q0 + r;
    const float negv = *negp;

    float m = -1e30f;
#pragma unroll
    for (int i = 0; i < 8; ++i) {
        const int k = i * 64 + c64;      // coalesced
        float sc = scores[q * NK + k];
        if (!mask[q * NK + k]) sc += negv;
        prob[r][k] = sc;
        m = fmaxf(m, sc);
    }
    red[r][c64] = m;
    __syncthreads();
    for (int s = 32; s > 0; s >>= 1) {
        if (c64 < s) red[r][c64] = fmaxf(red[r][c64], red[r][c64 + s]);
        __syncthreads();
    }
    const float mx = red[r][0];
    __syncthreads();

    float sum = 0.0f;
#pragma unroll
    for (int i = 0; i < 8; ++i) {
        const int k = i * 64 + c64;
        const float e = __expf(prob[r][k] - mx);
        prob[r][k] = e;
        sum += e;
    }
    red[r][c64] = sum;
    __syncthreads();
    for (int s = 32; s > 0; s >>= 1) {
        if (c64 < s) red[r][c64] += red[r][c64 + s];
        __syncthreads();
    }
    const float inv = 1.0f / red[r][0];
    __syncthreads();
#pragma unroll
    for (int i = 0; i < 8; ++i) prob[r][i * 64 + c64] *= inv;
    __syncthreads();

    // message[d, q0+rr] = sum_k prob[rr][k] * source[d, k]
    // one source-row read (L2-resident) amortized over 4 output columns
    const int d = tid;                   // 0..255
    const float4* src4 = (const float4*)(source + d * NK);
    float acc[4] = {0.0f, 0.0f, 0.0f, 0.0f};
#pragma unroll 2
    for (int k4 = 0; k4 < NK / 4; ++k4) {
        const float4 sv = src4[k4];
#pragma unroll
        for (int rr = 0; rr < 4; ++rr) {
            acc[rr] += sv.x * prob[rr][4 * k4 + 0] + sv.y * prob[rr][4 * k4 + 1] +
                       sv.z * prob[rr][4 * k4 + 2] + sv.w * prob[rr][4 * k4 + 3];
        }
    }
#pragma unroll
    for (int rr = 0; rr < 4; ++rr) message[d * NQ + (q0 + rr)] = acc[rr];
}

// ------------- Kernel D1: h2 = relu(mW1 @ [x;message] + mb1), 4 rows -------
__global__ __launch_bounds__(256) void merge_h_kernel(
    const float* __restrict__ mW1, const float* __restrict__ mb1,
    const float* __restrict__ x, const float* __restrict__ message,
    float* __restrict__ h2) {
    const int c0 = blockIdx.x * 4;       // grid 128
    const int tq = threadIdx.x;
    float a0[4], a1[4];
#pragma unroll
    for (int r = 0; r < 4; ++r) { a0[r] = mb1[c0 + r]; a1[r] = mb1[c0 + r]; }
#pragma unroll 2
    for (int i = 0; i < DD; ++i) {
        const float xv0 = x[i * NQ + tq];
        const float xv1 = x[i * NQ + tq + 256];
#pragma unroll
        for (int r = 0; r < 4; ++r) {
            const float wv = mW1[(c0 + r) * 512 + i];
            a0[r] += wv * xv0;
            a1[r] += wv * xv1;
        }
    }
#pragma unroll 2
    for (int i = 0; i < DD; ++i) {
        const float mv0 = message[i * NQ + tq];
        const float mv1 = message[i * NQ + tq + 256];
#pragma unroll
        for (int r = 0; r < 4; ++r) {
            const float wv = mW1[(c0 + r) * 512 + DD + i];
            a0[r] += wv * mv0;
            a1[r] += wv * mv1;
        }
    }
#pragma unroll
    for (int r = 0; r < 4; ++r) {
        h2[(c0 + r) * NQ + tq]       = fmaxf(a0[r], 0.0f);
        h2[(c0 + r) * NQ + tq + 256] = fmaxf(a1[r], 0.0f);
    }
}

// ------------------- Kernel D2: out = mW2 @ h2 + mb2, 4 rows ---------------
__global__ __launch_bounds__(256) void merge_out_kernel(
    const float* __restrict__ mW2, const float* __restrict__ mb2,
    const float* __restrict__ h2, float* __restrict__ out) {
    const int o0 = blockIdx.x * 4;       // grid 64
    const int tq = threadIdx.x;
    float a0[4], a1[4];
#pragma unroll
    for (int r = 0; r < 4; ++r) { a0[r] = mb2[o0 + r]; a1[r] = mb2[o0 + r]; }
#pragma unroll 2
    for (int c = 0; c < 512; ++c) {
        const float hv0 = h2[c * NQ + tq];
        const float hv1 = h2[c * NQ + tq + 256];
#pragma unroll
        for (int r = 0; r < 4; ++r) {
            const float wv = mW2[(o0 + r) * 512 + c];
            a0[r] += wv * hv0;
            a1[r] += wv * hv1;
        }
    }
#pragma unroll
    for (int r = 0; r < 4; ++r) {
        out[(o0 + r) * NQ + tq]       = a0[r];
        out[(o0 + r) * NQ + tq + 256] = a1[r];
    }
}

// ---------------------------------------------------------------------------
extern "C" void kernel_launch(void* const* d_in, const int* in_sizes, int n_in,
                              void* d_out, int out_size, void* d_ws, size_t ws_size,
                              hipStream_t stream) {
    const float*         x      = (const float*)d_in[0];
    const float*         source = (const float*)d_in[1];
    const float*         dist   = (const float*)d_in[2];
    const unsigned char* mask   = (const unsigned char*)d_in[3];
    const float*         aW1    = (const float*)d_in[4];
    const float*         ab1    = (const float*)d_in[5];
    const float*         aW2    = (const float*)d_in[6];
    const float*         ab2    = (const float*)d_in[7];
    const float*         mW1    = (const float*)d_in[8];
    const float*         mb1    = (const float*)d_in[9];
    const float*         mW2    = (const float*)d_in[10];
    const float*         mb2    = (const float*)d_in[11];

    float* out    = (float*)d_out;        // (1, D, NQ)    = 131072 floats
    float* scores = out + DD * NQ;        // (1, NQ, NK)   = 262144 floats

    // workspace layout (floats): U | V | message | h2 | minval  (~2.6 MB)
    float* U       = (float*)d_ws;
    float* V       = U + DD * NQ;
    float* message = V + DD * NK;
    float* h2      = message + DD * NQ;
    float* minval  = h2 + 2 * DD * NQ;

    uv_kernel<<<128, 256, 0, stream>>>(aW1, ab1, x, source, U, V);
    mlp_scores_kernel<<<512, 256, 0, stream>>>(aW1, aW2, ab2, dist, U, V, scores);
    min_reduce_kernel<<<1, 1024, 0, stream>>>(scores, minval);
    softmax_message_kernel<<<NQ / 4, 256, 0, stream>>>(scores, mask, minval, source, message);
    merge_h_kernel<<<128, 256, 0, stream>>>(mW1, mb1, x, message, h2);
    merge_out_kernel<<<64, 256, 0, stream>>>(mW2, mb2, h2, out);
}
